// Network_20109036880204
// MI455X (gfx1250) — compile-verified, matrix operand order checked by reading, not run on hardware
//
#include <hip/hip_runtime.h>
#include <hip/hip_bf16.h>
#include <stdint.h>

// ---------------------------------------------------------------------------
// NeuroFlame rate network, 50 steps of:
//   STP elementwise -> (32x8192)x(8192x8192) GEMM via bf16 WMMA -> rate update
// Wab converted once to bf16 (128MB, L2-resident); f32 accumulation.
// GEMM: each wave owns a 32(M) x 32(N) output tile = 2x2 WMMA grid, so A
// fragments are reused across n-tiles and B fragments across m-tiles
// (4 WMMAs per 8 loads). K split 8 ways into separate partial buffers
// (bitwise-deterministic), reduced in the fused epilogue.
// ---------------------------------------------------------------------------

typedef __attribute__((ext_vector_type(16))) __bf16 v16bf;
typedef __attribute__((ext_vector_type(8)))  __bf16 v8bf;
typedef __attribute__((ext_vector_type(8)))  float  v8f;
typedef __attribute__((ext_vector_type(4)))  float  v4f;

#define NN      8192
#define BB      32
#define BN      (BB * NN)          // 262144 state elements
#define NSTEPS  50                 // reference n_steps (device scalar; fixed by setup)
#define KSPLIT  8
#define KCHUNK  (NN / KSPLIT)      // 1024

// simulation constants
#define DTc      0.01f
#define USEc     0.03f
#define DT_FAC   0.01f             // DT / TAU_FAC
#define DT_REC   0.04f             // DT / TAU_REC

__device__ __forceinline__ __bf16 f32_to_bf16_rne(float f) {
  union { float f; uint32_t u; } v; v.f = f;
  uint32_t u = v.u;
  uint32_t r = (u + 0x7FFFu + ((u >> 16) & 1u)) >> 16;
  union { uint16_t s; __bf16 b; } o; o.s = (uint16_t)r;
  return o.b;
}

// --- one-time weight conversion: f32 (256MB) -> bf16 (128MB, L2-resident) ---
__global__ __launch_bounds__(256) void cvt_w_kernel(const float* __restrict__ w,
                                                    __bf16* __restrict__ wbf) {
  size_t i = ((size_t)blockIdx.x * 256 + threadIdx.x) * 4;
  v4f x = *(const v4f*)(w + i);
  __bf16 o0 = f32_to_bf16_rne(x.x);
  __bf16 o1 = f32_to_bf16_rne(x.y);
  __bf16 o2 = f32_to_bf16_rne(x.z);
  __bf16 o3 = f32_to_bf16_rne(x.w);
  __bf16* p = wbf + i;
  p[0] = o0; p[1] = o1; p[2] = o2; p[3] = o3;
}

// --- per-step STP update; emits bf16 activity s = u_n * x_n * r ---
__global__ __launch_bounds__(256) void stp_kernel(const float* __restrict__ r,
                                                  float* __restrict__ u,
                                                  float* __restrict__ x,
                                                  __bf16* __restrict__ sbf) {
  int i = blockIdx.x * 256 + threadIdx.x;
  float rv = r[i], uv = u[i], xv = x[i];
  float un = uv + DT_FAC * (USEc - uv) + USEc * (1.0f - uv) * rv * DTc;
  float xn = xv + DT_REC * (1.0f - xv) - un * xv * rv * DTc;
  u[i] = un;
  x[i] = xn;
  sbf[i] = f32_to_bf16_rne(un * xn * rv);
}

// --- GEMM: y[b,j] = sum_i s[b,i] * W[j,i], bf16 WMMA, f32 accumulate ------
// grid (64, KSPLIT) x 128 threads (4 waves). wave -> 32x32 output tile
// (2 n-tiles x 2 m-tiles), K-chunk of 1024.
__global__ __launch_bounds__(128) void gemm_kernel(const __bf16* __restrict__ sbf,
                                                   const __bf16* __restrict__ wbf,
                                                   float* __restrict__ ypart) {
  const int lane = threadIdx.x & 31;
  const int wave = threadIdx.x >> 5;
  const int lo16 = lane & 15;
  const int lhib = (lane >> 4) << 3;                 // 0 or 8 (K sub-offset)
  const int n0   = (blockIdx.x * 4 + wave) * 32;     // 2 n-tiles per wave
  const int kbeg = blockIdx.y * KCHUNK;
  const int kend = kbeg + KCHUNK;

  const __bf16* arow0 = sbf + (size_t)lo16 * NN;          // A tile 0: m = lo16
  const __bf16* arow1 = sbf + (size_t)(lo16 + 16) * NN;   // A tile 1: m = lo16+16
  const __bf16* brow0 = wbf + (size_t)(n0 + lo16) * NN;        // B tile 0
  const __bf16* brow1 = wbf + (size_t)(n0 + 16 + lo16) * NN;   // B tile 1

  v8f c00 = {};   // m-tile 0, n-tile 0
  v8f c10 = {};   // m-tile 1, n-tile 0
  v8f c01 = {};   // m-tile 0, n-tile 1
  v8f c11 = {};   // m-tile 1, n-tile 1

  for (int k0 = kbeg; k0 < kend; k0 += 32) {
    // each operand fragment: two contiguous 16B runs per lane (K layout §7.12.2)
    v8bf a0l = *(const v8bf*)(arow0 + k0 + lhib);
    v8bf a0h = *(const v8bf*)(arow0 + k0 + 16 + lhib);
    v8bf a1l = *(const v8bf*)(arow1 + k0 + lhib);
    v8bf a1h = *(const v8bf*)(arow1 + k0 + 16 + lhib);
    v8bf b0l = *(const v8bf*)(brow0 + k0 + lhib);
    v8bf b0h = *(const v8bf*)(brow0 + k0 + 16 + lhib);
    v8bf b1l = *(const v8bf*)(brow1 + k0 + lhib);
    v8bf b1h = *(const v8bf*)(brow1 + k0 + 16 + lhib);

    v16bf a0 = __builtin_shufflevector(a0l, a0h, 0,1,2,3,4,5,6,7,8,9,10,11,12,13,14,15);
    v16bf a1 = __builtin_shufflevector(a1l, a1h, 0,1,2,3,4,5,6,7,8,9,10,11,12,13,14,15);
    v16bf b0 = __builtin_shufflevector(b0l, b0h, 0,1,2,3,4,5,6,7,8,9,10,11,12,13,14,15);
    v16bf b1 = __builtin_shufflevector(b1l, b1h, 0,1,2,3,4,5,6,7,8,9,10,11,12,13,14,15);

    c00 = __builtin_amdgcn_wmma_f32_16x16x32_bf16(false, a0, false, b0, (short)0, c00, false, false);
    c01 = __builtin_amdgcn_wmma_f32_16x16x32_bf16(false, a0, false, b1, (short)0, c01, false, false);
    c10 = __builtin_amdgcn_wmma_f32_16x16x32_bf16(false, a1, false, b0, (short)0, c10, false, false);
    c11 = __builtin_amdgcn_wmma_f32_16x16x32_bf16(false, a1, false, b1, (short)0, c11, false, false);
  }

  // C/D layout: m = vgpr + (lane>=16 ? 8 : 0), n = lane%16
  float* yp = ypart + (size_t)blockIdx.y * BN;
  const int na    = n0 + lo16;
  const int nb    = n0 + 16 + lo16;
  const int mbase = (lane >> 4) << 3;
#pragma unroll
  for (int v = 0; v < 8; ++v) {
    int m = v + mbase;
    yp[(size_t)m * NN + na]        = c00[v];
    yp[(size_t)(m + 16) * NN + na] = c10[v];
    yp[(size_t)m * NN + nb]        = c01[v];
    yp[(size_t)(m + 16) * NN + nb] = c11[v];
  }
}

// --- epilogue: deterministic K-partial reduction + rec/rate update ---------
__global__ __launch_bounds__(256) void epi_kernel(const float* __restrict__ yp,
                                                  const float* __restrict__ ff,
                                                  const float* __restrict__ e_syn,
                                                  const float* __restrict__ dt_syn,
                                                  const float* __restrict__ e_tau,
                                                  const float* __restrict__ dt_tau,
                                                  float* __restrict__ rec,
                                                  float* __restrict__ r) {
  int i = blockIdx.x * 256 + threadIdx.x;
  int n = i & (NN - 1);
  float y = 0.0f;
#pragma unroll
  for (int p = 0; p < KSPLIT; ++p)
    y += yp[(size_t)p * BN + i];               // fixed order: deterministic
  float rec_n = e_syn[n] * rec[i] + dt_syn[n] * y;
  float net   = ff[i] + rec_n;                 // THRESH == 0
  float relu  = net > 0.0f ? net : 0.0f;
  float rn    = e_tau[n] * r[i] + dt_tau[n] * relu;
  rec[i] = rec_n;
  r[i]   = rn;
}

extern "C" void kernel_launch(void* const* d_in, const int* in_sizes, int n_in,
                              void* d_out, int out_size, void* d_ws, size_t ws_size,
                              hipStream_t stream) {
  const float* rates     = (const float*)d_in[0];
  const float* rec_in    = (const float*)d_in[1];
  const float* ff        = (const float*)d_in[2];
  const float* Wab       = (const float*)d_in[3];
  const float* u_in      = (const float*)d_in[4];
  const float* x_in      = (const float*)d_in[5];
  const float* e_tau     = (const float*)d_in[6];
  const float* dt_tau    = (const float*)d_in[7];
  const float* e_syn     = (const float*)d_in[8];
  const float* dt_syn    = (const float*)d_in[9];
  (void)in_sizes; (void)n_in; (void)out_size; (void)ws_size;

  // workspace carve-up (all offsets 256B aligned)
  char* ws = (char*)d_ws;
  __bf16* wbf   = (__bf16*)ws;                      ws += (size_t)NN * NN * 2;   // 128 MB
  float*  r     = (float*)ws;                       ws += (size_t)BN * 4;
  float*  rec   = (float*)ws;                       ws += (size_t)BN * 4;
  float*  u     = (float*)ws;                       ws += (size_t)BN * 4;
  float*  x     = (float*)ws;                       ws += (size_t)BN * 4;
  float*  ypart = (float*)ws;                       ws += (size_t)BN * 4 * KSPLIT;
  __bf16* sbf   = (__bf16*)ws;                      ws += (size_t)BN * 2;

  // one-time per call: bf16 weights + state copies (inputs never mutated)
  cvt_w_kernel<<<((size_t)NN * NN) / (256 * 4), 256, 0, stream>>>(Wab, wbf);
  hipMemcpyAsync(r,   rates,  (size_t)BN * 4, hipMemcpyDeviceToDevice, stream);
  hipMemcpyAsync(rec, rec_in, (size_t)BN * 4, hipMemcpyDeviceToDevice, stream);
  hipMemcpyAsync(u,   u_in,   (size_t)BN * 4, hipMemcpyDeviceToDevice, stream);
  hipMemcpyAsync(x,   x_in,   (size_t)BN * 4, hipMemcpyDeviceToDevice, stream);

  dim3 gemm_grid(NN / 128, KSPLIT);   // 4 waves/block * 32 cols = 128 cols/block
  for (int t = 0; t < NSTEPS; ++t) {
    stp_kernel<<<BN / 256, 256, 0, stream>>>(r, u, x, sbf);
    gemm_kernel<<<gemm_grid, 128, 0, stream>>>(sbf, wbf, ypart);
    epi_kernel<<<BN / 256, 256, 0, stream>>>(ypart, ff, e_syn, dt_syn,
                                             e_tau, dt_tau, rec, r);
  }

  hipMemcpyAsync(d_out, r, (size_t)BN * 4, hipMemcpyDeviceToDevice, stream);
}
